// combined_model_60086592471726
// MI455X (gfx1250) — compile-verified, matrix operand order checked by reading, not run on hardware
//
#include <hip/hip_runtime.h>
#include <cstddef>

// ---------------------------------------------------------------------------
// Types for CDNA5 WMMA (wave32): 16x16x32 bf16 -> f32
// ---------------------------------------------------------------------------
typedef __attribute__((ext_vector_type(16))) __bf16        v16bf;
typedef __attribute__((ext_vector_type(8)))  float         v8f;
typedef __attribute__((ext_vector_type(4)))  unsigned int  v4u;
typedef __attribute__((ext_vector_type(2)))  unsigned int  v2u;

static constexpr int B_SZ = 2048;
static constexpr int T_SZ = 64;
static constexpr int H_SZ = 64;    // LSTM hidden
static constexpr int G_SZ = 256;   // 4*H gates

// ---------------------------------------------------------------------------
// Fast transcendentals (hardware v_tanh_f32 confirmed present in round 2)
// ---------------------------------------------------------------------------
__device__ inline float fast_rcp(float x) {
#if __has_builtin(__builtin_amdgcn_rcpf)
  return __builtin_amdgcn_rcpf(x);
#else
  return 1.f / x;
#endif
}

#if __has_builtin(__builtin_amdgcn_tanhf)
#define ATHENA_HW_TANH 1
__device__ inline float fast_tanh(float x) { return __builtin_amdgcn_tanhf(x); }
#elif __has_builtin(__builtin_amdgcn_tanh_f32)
#define ATHENA_HW_TANH 1
__device__ inline float fast_tanh(float x) { return __builtin_amdgcn_tanh_f32(x); }
#else
#define ATHENA_HW_TANH 0
__device__ inline float fast_tanh(float x) {
  float e = __expf(-2.f * x);
  return (1.f - e) * fast_rcp(1.f + e);
}
#endif

__device__ inline float fast_sigm(float x) {
#if ATHENA_HW_TANH
  return __builtin_fmaf(0.5f, fast_tanh(0.5f * x), 0.5f);
#else
  return fast_rcp(1.f + __expf(-x));
#endif
}

// ---------------------------------------------------------------------------
// Async global->LDS copy (CDNA5 Tensor/async path, ASYNCcnt-tracked).
// LDS destination address = low 32 bits of the flat shared address
// (ISA: LDS aperture maps addr[31:0] -> LDS byte address).
// ---------------------------------------------------------------------------
__device__ inline void async_copy_b128(unsigned lds_off, const void* gptr) {
  asm volatile("global_load_async_to_lds_b128 %0, %1, off"
               :: "v"(lds_off), "v"(gptr)
               : "memory");
}

__device__ inline void wait_async0() {
#if __has_builtin(__builtin_amdgcn_s_wait_asynccnt)
  __builtin_amdgcn_s_wait_asynccnt(0);
#else
  asm volatile("s_wait_asynccnt 0x0" ::: "memory");
#endif
}

// ---------------------------------------------------------------------------
// Input MLP: x[B,T,2] -> relu(x@w1^T+b1)@w2^T+b2 -> bf16 [B,T,16]
// ---------------------------------------------------------------------------
__global__ __launch_bounds__(256) void mlp_in_kernel(
    const float* __restrict__ x,
    const float* __restrict__ w1, const float* __restrict__ b1,
    const float* __restrict__ w2, const float* __restrict__ b2,
    __bf16* __restrict__ out) {
  __shared__ float sw1[32], sb1[16], sw2[256], sb2[16];
  const int tid = threadIdx.x;
  if (tid < 32) sw1[tid] = w1[tid];
  if (tid < 16) { sb1[tid] = b1[tid]; sb2[tid] = b2[tid]; }
  if (tid < 256) sw2[tid] = w2[tid];
  __syncthreads();

  const int g = blockIdx.x * 256 + tid;      // 0 .. B*T-1
  const float x0 = x[(size_t)2 * g];
  const float x1 = x[(size_t)2 * g + 1];
  float hid[16];
#pragma unroll
  for (int j = 0; j < 16; ++j) {
    float s = sw1[2 * j] * x0 + sw1[2 * j + 1] * x1 + sb1[j];
    hid[j] = fmaxf(s, 0.f);
  }
#pragma unroll
  for (int j = 0; j < 16; ++j) {
    float s = sb2[j];
#pragma unroll
    for (int k = 0; k < 16; ++k) s += sw2[j * 16 + k] * hid[k];
    out[(size_t)g * 16 + j] = (__bf16)s;
  }
}

// ---------------------------------------------------------------------------
// Issue async staging of one 16-step chunk into an LDS x-tile buffer.
// Each 8-half A-fragment group is contiguous in memory -> one async b128
// copy per lane per iteration (groups beyond IN_DIM stay pre-zeroed).
// ---------------------------------------------------------------------------
template <int IN_DIM, int ICH>
__device__ inline void issue_stage(const __bf16* __restrict__ in, int Bb, int t0,
                                   unsigned lds_base, int tid) {
  const int nGrp = 16 * ICH * 64;          // groups of 8 halfs
  for (int hidx = tid; hidx < nGrp; hidx += 256) {
    int half = hidx & 1;                   // 0 -> e0..7, 1 -> e8..15
    int l    = (hidx >> 1) & 31;
    int rest = hidx >> 6;
    int c    = rest % ICH;
    int tc   = rest / ICH;
    int M    = l & 15;
    bool lh  = l >= 16;
    int k0 = 32 * c + (half ? (lh ? 24 : 16) : (lh ? 8 : 0));
    if (k0 + 8 <= IN_DIM) {
      const __bf16* g = in + ((size_t)(Bb + M) * T_SZ + (t0 + tc)) * IN_DIM + k0;
      async_copy_b128(lds_base + (unsigned)hidx * 16u, (const void*)g);
    }
  }
}

// ---------------------------------------------------------------------------
// One LSTM layer, fully fused per 16-row batch tile.
//   IN_DIM = input width (16 for layer 0, 64 otherwise)
//   ICH    = number of 32-wide K chunks covering the input (1 or 2)
//
// 8 waves; wave w owns gate N-tiles {w, w+8}:
//   waves 0..3 -> i-gates (tiles 0..3) and g-gates (tiles 8..11)
//   waves 4..7 -> f-gates (tiles 4..7) and o-gates (tiles 12..15)
// x-tile staging is double-buffered with async global->LDS copies so global
// latency overlaps the WMMA scan of the current chunk.
// ---------------------------------------------------------------------------
template <int IN_DIM, int ICH>
__global__ __launch_bounds__(256) void lstm_layer_kernel(
    const __bf16* __restrict__ in,   // [B,T,IN_DIM] bf16
    __bf16* __restrict__ out,        // [B,T,64] bf16
    const float* __restrict__ w_ih,  // [256, IN_DIM]
    const float* __restrict__ w_hh,  // [256, 64]
    const float* __restrict__ b_ih,  // [256]
    const float* __restrict__ b_hh)  // [256]
{
  constexpr int BUF = 16 * ICH * 512;                    // halfs per x buffer
  extern __shared__ char smem[];
  __bf16* x_tile = (__bf16*)smem;                        // 2 * BUF halfs
  __bf16* h_sh   = x_tile + (size_t)2 * BUF;             // 1024 halfs (A-fragment order)
  float*  ig_sh  = (float*)(h_sh + 1024);                // 16 x 64 f32
  __bf16* h_lin  = (__bf16*)(ig_sh + 1024);              // 16 x 64 bf16 (linear)

  const int tid  = threadIdx.x;
  const int lane = tid & 31;
  const int wv   = tid >> 5;
  const int Bb   = blockIdx.x * 16;   // batch tile base
  const int N    = lane & 15;
  const bool hi  = lane >= 16;

  for (int i = tid; i < 1024; i += 256) h_sh[i] = (__bf16)0.f;   // h0 = 0

  // Pre-zero groups never covered by async loads (layer 0: K >= IN_DIM pad)
  if (IN_DIM < 32 * ICH) {
    const int nGrp = 16 * ICH * 64;
    for (int hidx = tid; hidx < nGrp; hidx += 256) {
      int half = hidx & 1;
      int l    = (hidx >> 1) & 31;
      int c    = (hidx >> 6) % ICH;
      bool lh  = l >= 16;
      int k0 = 32 * c + (half ? (lh ? 24 : 16) : (lh ? 8 : 0));
      if (k0 + 8 > IN_DIM) {
        v4u z = {0u, 0u, 0u, 0u};
        *(v4u*)(x_tile + (size_t)hidx * 8) = z;
        *(v4u*)(x_tile + (size_t)BUF + (size_t)hidx * 8) = z;
      }
    }
  }

  // ---- Build resident B fragments (weights, f32 -> bf16) --------------------
  const int tileA = wv, tileB = wv + 8;
  v16bf BxA[ICH], BxB[ICH], BhA[2], BhB[2];
  {
    const float* rxA = w_ih + (size_t)(16 * tileA + N) * IN_DIM;
    const float* rxB = w_ih + (size_t)(16 * tileB + N) * IN_DIM;
#pragma unroll
    for (int c = 0; c < ICH; ++c) {
#pragma unroll
      for (int e = 0; e < 16; ++e) {
        int k = 32 * c + (e < 8 ? (hi ? e + 8 : e) : (hi ? e + 16 : e + 8));
        BxA[c][e] = (__bf16)((k < IN_DIM) ? rxA[k] : 0.f);
        BxB[c][e] = (__bf16)((k < IN_DIM) ? rxB[k] : 0.f);
      }
    }
    const float* rhA = w_hh + (size_t)(16 * tileA + N) * H_SZ;
    const float* rhB = w_hh + (size_t)(16 * tileB + N) * H_SZ;
#pragma unroll
    for (int c = 0; c < 2; ++c) {
#pragma unroll
      for (int e = 0; e < 16; ++e) {
        int k = 32 * c + (e < 8 ? (hi ? e + 8 : e) : (hi ? e + 16 : e + 8));
        BhA[c][e] = (__bf16)rhA[k];
        BhB[c][e] = (__bf16)rhB[k];
      }
    }
  }
  const float bA = b_ih[16 * tileA + N] + b_hh[16 * tileA + N];
  const float bB = b_ih[16 * tileB + N] + b_hh[16 * tileB + N];

  float cst[8];
#pragma unroll
  for (int j = 0; j < 8; ++j) cst[j] = 0.f;   // cell state (waves 4..7)

  const unsigned x_base = (unsigned)(size_t)x_tile;   // LDS byte address
  // Kick off chunk 0 staging
  issue_stage<IN_DIM, ICH>(in, Bb, 0, x_base, tid);

  // ---- Scan over T in chunks of 16 steps ------------------------------------
  for (int chunk = 0; chunk < T_SZ / 16; ++chunk) {
    wait_async0();          // my slice of this chunk's async copies done
    __syncthreads();        // everyone's copies visible in LDS
    if (chunk + 1 < T_SZ / 16)   // prefetch next chunk into the other buffer
      issue_stage<IN_DIM, ICH>(in, Bb, (chunk + 1) * 16,
                               x_base + (unsigned)((chunk + 1) & 1) * (BUF * 2u), tid);
    const __bf16* xb = x_tile + (size_t)(chunk & 1) * BUF;

    for (int tc = 0; tc < 16; ++tc) {
      const int t = chunk * 16 + tc;
      // A fragments (uniform control flow; EXEC all-1s for WMMA)
      v16bf hA0 = *(const v16bf*)(h_sh + (size_t)(0 * 32 + lane) * 16);
      v16bf hA1 = *(const v16bf*)(h_sh + (size_t)(1 * 32 + lane) * 16);
      v16bf xA[ICH];
#pragma unroll
      for (int c = 0; c < ICH; ++c)
        xA[c] = *(const v16bf*)(xb + (size_t)((tc * ICH + c) * 32 + lane) * 16);

      v8f accA = {bA, bA, bA, bA, bA, bA, bA, bA};
      v8f accB = {bB, bB, bB, bB, bB, bB, bB, bB};
#pragma unroll
      for (int c = 0; c < ICH; ++c) {
        accA = __builtin_amdgcn_wmma_f32_16x16x32_bf16(false, xA[c], false, BxA[c],
                                                       (short)0, accA, false, false);
        accB = __builtin_amdgcn_wmma_f32_16x16x32_bf16(false, xA[c], false, BxB[c],
                                                       (short)0, accB, false, false);
      }
      accA = __builtin_amdgcn_wmma_f32_16x16x32_bf16(false, hA0, false, BhA[0],
                                                     (short)0, accA, false, false);
      accA = __builtin_amdgcn_wmma_f32_16x16x32_bf16(false, hA1, false, BhA[1],
                                                     (short)0, accA, false, false);
      accB = __builtin_amdgcn_wmma_f32_16x16x32_bf16(false, hA0, false, BhB[0],
                                                     (short)0, accB, false, false);
      accB = __builtin_amdgcn_wmma_f32_16x16x32_bf16(false, hA1, false, BhB[1],
                                                     (short)0, accB, false, false);

      float fo_f[8], fo_o[8];
      if (wv >= 4) {
        // f/o sigmoids don't depend on the i*g exchange: compute pre-barrier
#pragma unroll
        for (int j = 0; j < 8; ++j) {
          fo_f[j] = fast_sigm(accA[j]);
          fo_o[j] = fast_sigm(accB[j]);
        }
      } else {
        const int u = 16 * wv + N;
#pragma unroll
        for (int j = 0; j < 8; ++j) {
          int M = j + (hi ? 8 : 0);
          ig_sh[M * 64 + u] = fast_sigm(accA[j]) * fast_tanh(accB[j]);
        }
      }
      __syncthreads();
      if (wv >= 4) {
        const int u = 16 * (wv - 4) + N;
#pragma unroll
        for (int j = 0; j < 8; ++j) {
          int M = j + (hi ? 8 : 0);
          float cc = __builtin_fmaf(fo_f[j], cst[j], ig_sh[M * 64 + u]);
          cst[j]   = cc;
          float hv = fo_o[j] * fast_tanh(cc);
          __bf16 hb = (__bf16)hv;
          h_lin[M * 64 + u] = hb;
          // republish h into A-fragment swizzle order for the next step
          int c  = u >> 5;
          int km = u & 31;
          int lp = M + (((km & 15) >= 8) ? 16 : 0);
          int e  = (km & 7) + ((km >= 16) ? 8 : 0);
          h_sh[(size_t)(c * 32 + lp) * 16 + e] = hb;
        }
      }
      __syncthreads();
      // Cooperative vectorized store of h(t): 256 threads x 8 bytes
      {
        int M  = tid >> 4;          // 0..15
        int k0 = (tid & 15) * 4;    // 0..60
        *(v2u*)(out + ((size_t)(Bb + M) * T_SZ + t) * H_SZ + k0) =
            *(const v2u*)(h_lin + M * 64 + k0);
      }
    }
  }
}

// ---------------------------------------------------------------------------
// Output MLP: h[B,T,64] bf16 -> relu(h@wo1^T+bo1)@wo2^T+bo2 -> f32 [B,T,4]
// ---------------------------------------------------------------------------
__global__ __launch_bounds__(256) void mlp_out_kernel(
    const __bf16* __restrict__ h,
    const float* __restrict__ wo1, const float* __restrict__ bo1,
    const float* __restrict__ wo2, const float* __restrict__ bo2,
    float* __restrict__ out) {
  __shared__ float s1[32 * 64], sb1[32], s2[4 * 32], sb2[4];
  const int tid = threadIdx.x;
  for (int i = tid; i < 2048; i += 256) s1[i] = wo1[i];
  if (tid < 32) sb1[tid] = bo1[tid];
  if (tid < 128) s2[tid] = wo2[tid];
  if (tid < 4) sb2[tid] = bo2[tid];
  __syncthreads();

  const int g = blockIdx.x * 256 + tid;      // 0 .. B*T-1
  const __bf16* hr = h + (size_t)g * 64;
  float hid[32];
#pragma unroll
  for (int j = 0; j < 32; ++j) hid[j] = sb1[j];
  for (int k = 0; k < 64; ++k) {
    float xk = (float)hr[k];
#pragma unroll
    for (int j = 0; j < 32; ++j) hid[j] += xk * s1[j * 64 + k];
  }
  float o[4] = {sb2[0], sb2[1], sb2[2], sb2[3]};
#pragma unroll
  for (int j = 0; j < 32; ++j) {
    float hj = fmaxf(hid[j], 0.f);
#pragma unroll
    for (int m = 0; m < 4; ++m) o[m] += hj * s2[m * 32 + j];
  }
#pragma unroll
  for (int m = 0; m < 4; ++m) out[(size_t)g * 4 + m] = o[m];
}

// ---------------------------------------------------------------------------
// Host-side orchestration
// ---------------------------------------------------------------------------
extern "C" void kernel_launch(void* const* d_in, const int* in_sizes, int n_in,
                              void* d_out, int out_size, void* d_ws, size_t ws_size,
                              hipStream_t stream) {
  (void)in_sizes; (void)n_in; (void)out_size; (void)ws_size;
  const float* x     = (const float*)d_in[0];
  const float* w1    = (const float*)d_in[1];
  const float* b1    = (const float*)d_in[2];
  const float* w2    = (const float*)d_in[3];
  const float* b2    = (const float*)d_in[4];
  const float* w_ih0 = (const float*)d_in[5];
  const float* w_hh0 = (const float*)d_in[6];
  const float* b_ih0 = (const float*)d_in[7];
  const float* b_hh0 = (const float*)d_in[8];
  const float* w_ih  = (const float*)d_in[9];
  const float* w_hh  = (const float*)d_in[10];
  const float* b_ih  = (const float*)d_in[11];
  const float* b_hh  = (const float*)d_in[12];
  const float* wo1   = (const float*)d_in[13];
  const float* bo1   = (const float*)d_in[14];
  const float* wo2   = (const float*)d_in[15];
  const float* bo2   = (const float*)d_in[16];

  // Workspace layout (bf16 activations):
  //   bufA: [B,T,16]  = 4 MB   (input-MLP output)
  //   bufB: [B,T,64]  = 16 MB  (ping)
  //   bufC: [B,T,64]  = 16 MB  (pong)
  char* ws = (char*)d_ws;
  __bf16* bufA = (__bf16*)ws;
  __bf16* bufB = (__bf16*)(ws + ((size_t)4 << 20));
  __bf16* bufC = (__bf16*)(ws + ((size_t)20 << 20));

  const int nBT_blocks   = (B_SZ * T_SZ) / 256;   // 512
  const int nLSTM_blocks = B_SZ / 16;             // 128

  mlp_in_kernel<<<nBT_blocks, 256, 0, stream>>>(x, w1, b1, w2, b2, bufA);

  // smem = 2*x_buf + h_sh(2KB) + ig_sh(4KB) + h_lin(2KB)
  constexpr size_t smem1 = (size_t)2 * 16 * 1 * 512 * 2 + 2048 + 4096 + 2048; // 40960 B
  constexpr size_t smem2 = (size_t)2 * 16 * 2 * 512 * 2 + 2048 + 4096 + 2048; // 73728 B

  lstm_layer_kernel<16, 1><<<nLSTM_blocks, 256, smem1, stream>>>(
      bufA, bufB, w_ih0, w_hh0, b_ih0, b_hh0);

  const __bf16* cin = bufB;
  __bf16* cout = bufC;
  for (int l = 0; l < 7; ++l) {
    lstm_layer_kernel<64, 2><<<nLSTM_blocks, 256, smem2, stream>>>(
        cin, cout,
        w_ih + (size_t)l * G_SZ * H_SZ, w_hh + (size_t)l * G_SZ * H_SZ,
        b_ih + (size_t)l * G_SZ,        b_hh + (size_t)l * G_SZ);
    const __bf16* tmp = cout;
    cout = (__bf16*)cin;
    cin  = tmp;
  }

  mlp_out_kernel<<<nBT_blocks, 256, 0, stream>>>(cin, wo1, bo1, wo2, bo2,
                                                 (float*)d_out);
}
